// t5_42468636623267
// MI455X (gfx1250) — compile-verified, matrix operand order checked by reading, not run on hardware
//
#include <hip/hip_runtime.h>
#include <hip/hip_bf16.h>
#include <math.h>

#define SEQ    128
#define DMODEL 512
#define NHEADS 8
#define HDIM   64
#define EPS_F  1e-6f

typedef __attribute__((ext_vector_type(8)))  float  v8f;
typedef __attribute__((ext_vector_type(2)))  float  v2f;
typedef __attribute__((ext_vector_type(16))) __bf16 v16bf;

// Prefer exact f32 WMMA (V_WMMA_F32_16X16X4_F32); fall back to the
// codegen-confirmed bf16 WMMA (f32 accumulate) if the builtin is absent.
#if defined(__has_builtin)
#  if __has_builtin(__builtin_amdgcn_wmma_f32_16x16x4_f32)
#    define USE_F32_WMMA 1
#  endif
#endif
#ifndef USE_F32_WMMA
#  define USE_F32_WMMA 0
#endif

#if USE_F32_WMMA
typedef v2f FragT;
#define KSTEP 4
#else
typedef v16bf FragT;
#define KSTEP 32
#endif

// ---------------------------------------------------------------------------
// Fragment loader. Element (i, k) lives at base[(i0+i)*sI + (k0+k)*sK].
// For A-fragments i = M row; for B-fragments i = N column (layouts are
// symmetric over the K dimension per the ISA tables).
// ---------------------------------------------------------------------------
__device__ __forceinline__ FragT load_frag(const float* base, int sI, int sK,
                                           int i0, int k0, int lane) {
    int i  = lane & 15;
    int kh = lane >> 4;
    const float* p = base + (i0 + i) * sI;
#if USE_F32_WMMA
    // 32-bit A 16x4: VGPR0 = K0 (lanes 0-15) / K2 (lanes 16-31); VGPR1 = K1/K3
    FragT r;
    r.x = p[(k0 + 2 * kh + 0) * sK];
    r.y = p[(k0 + 2 * kh + 1) * sK];
    return r;
#else
    // 16-bit A 16x32: VGPR v<4 -> K = kh*8 + 2*(v%4)+t ; v>=4 -> +16
    FragT r;
#pragma unroll
    for (int e = 0; e < 16; ++e) {
        int v = e >> 1, t = e & 1;
        int k = ((v >> 2) << 4) + (kh << 3) + ((v & 3) << 1) + t;
        r[e] = (__bf16)p[(k0 + k) * sK];
    }
    return r;
#endif
}

__device__ __forceinline__ v8f wmma_step(FragT a, FragT b, v8f c) {
#if USE_F32_WMMA
    return __builtin_amdgcn_wmma_f32_16x16x4_f32(
        false, a, false, b, (short)0, c, false, false);
#else
    return __builtin_amdgcn_wmma_f32_16x16x32_bf16(
        false, a, false, b, (short)0, c, false, false);
#endif
}

// One 16x16 output tile, accumulating over KTOT.
template <int KTOT>
__device__ __forceinline__ v8f mma_tile(const float* A, int AsI, int AsK, int m0,
                                        const float* B, int BsI, int BsK, int n0,
                                        int lane) {
    v8f c = {};
    for (int k = 0; k < KTOT; k += KSTEP) {
        FragT a = load_frag(A, AsI, AsK, m0, k, lane);
        FragT b = load_frag(B, BsI, BsK, n0, k, lane);
        c = wmma_step(a, b, c);
    }
    return c;
}

// ---------------------------------------------------------------------------
// Kernel 1: T5 relative-position bias + key mask.  out idx = ((h*S)+i)*S + j
// ---------------------------------------------------------------------------
__global__ void relbias_kernel(const float* __restrict__ rbw,
                               const int* __restrict__ x_len,
                               float* __restrict__ rel_bias) {
    int idx = blockIdx.x * blockDim.x + threadIdx.x;
    if (idx >= NHEADS * SEQ * SEQ) return;
    int j    = idx & (SEQ - 1);
    int i    = (idx >> 7) & (SEQ - 1);
    int head = idx >> 14;

    int n   = j - i;                 // n = -(i - j)
    int ret = (n < 0) ? 16 : 0;
    n = (n < 0) ? -n : n;
    int bucket;
    if (n < 8) {
        bucket = ret + n;
    } else {
        float nf = (float)n;
        int vl = 8 + (int)(logf(nf * 0.125f) * (8.0f / logf(16.0f)));
        vl = (vl > 15) ? 15 : vl;
        bucket = ret + vl;
    }
    float b  = rbw[head * 32 + bucket];
    float mi = (i < x_len[0]) ? 0.0f : 1.0f;   // (1 - mask[i])
    rel_bias[idx] = b + mi * -1e9f;
}

// ---------------------------------------------------------------------------
// Kernel 2: embedding gather + RMSNorm + sequence mask. 128 threads / token.
// ---------------------------------------------------------------------------
__global__ void rmsnorm_kernel(const int* __restrict__ x,
                               const int* __restrict__ x_len,
                               const float* __restrict__ embed,
                               const float* __restrict__ scale,
                               float* __restrict__ h) {
    int s   = blockIdx.x;
    int tid = threadIdx.x;               // 0..127, 4 floats each
    long tok = (long)x[s];
    const float4* row = (const float4*)(embed + tok * DMODEL);
    float4 e = row[tid];
    float ss = e.x * e.x + e.y * e.y + e.z * e.z + e.w * e.w;
    for (int off = 16; off; off >>= 1) ss += __shfl_xor(ss, off, 32);
    __shared__ float red[4];
    if ((tid & 31) == 0) red[tid >> 5] = ss;
    __syncthreads();
    float var = (red[0] + red[1] + red[2] + red[3]) * (1.0f / DMODEL);
    float r   = rsqrtf(var + EPS_F);
    float msk = (s < x_len[0]) ? 1.0f : 0.0f;
    float4 sc = ((const float4*)scale)[tid];
    float4 o;
    o.x = e.x * r * sc.x * msk;
    o.y = e.y * r * sc.y * msk;
    o.z = e.z * r * sc.z * msk;
    o.w = e.w * r * sc.w * msk;
    ((float4*)(h + (long)s * DMODEL))[tid] = o;
}

// ---------------------------------------------------------------------------
// Kernel 3: q/k/v = h @ {qw,kw,vw}, stored head-split [head][s][d].
// 768 waves total: wave -> (matrix, 16x16 tile).
// ---------------------------------------------------------------------------
__global__ void qkv_kernel(const float* __restrict__ h,
                           const float* __restrict__ qw,
                           const float* __restrict__ kw,
                           const float* __restrict__ vw,
                           float* __restrict__ q,
                           float* __restrict__ k,
                           float* __restrict__ v) {
    int wave = (blockIdx.x * blockDim.x + threadIdx.x) >> 5;
    int lane = threadIdx.x & 31;
    int mat  = wave >> 8;                 // 0..2  (256 tiles per matrix)
    int t    = wave & 255;
    int mt   = t >> 5;                    // 8 M tiles
    int nt   = t & 31;                    // 32 N tiles
    const float* w    = (mat == 0) ? qw : (mat == 1) ? kw : vw;
    float*       outp = (mat == 0) ? q  : (mat == 1) ? k  : v;
    int m0 = mt * 16, n0 = nt * 16;

    // A: h[s][kk] (sI=512,sK=1).  B: w[kk][n] (sI=1, sK=512).
    v8f c = mma_tile<DMODEL>(h, DMODEL, 1, m0, w, 1, DMODEL, n0, lane);

    int n = lane & 15, half = lane >> 4;
    int col  = n0 + n;
    int head = col >> 6, dd = col & 63;
#pragma unroll
    for (int vv = 0; vv < 8; ++vv) {
        int s = m0 + vv + 8 * half;
        outp[head * (SEQ * HDIM) + s * HDIM + dd] = c[vv];
    }
}

// ---------------------------------------------------------------------------
// Kernel 4: attn_weights[h][qi][kj] = q.kT + rel_bias[h][kj][qi].
// 512 waves: (head, 8x8 tiles of 128x128).
// ---------------------------------------------------------------------------
__global__ void scores_kernel(const float* __restrict__ q,
                              const float* __restrict__ k,
                              const float* __restrict__ rel_bias,
                              float* __restrict__ attn_w) {
    int wave = (blockIdx.x * blockDim.x + threadIdx.x) >> 5;
    int lane = threadIdx.x & 31;
    int head = wave >> 6;
    int t    = wave & 63;
    int mt   = t >> 3, nt = t & 7;
    const float* qh = q + head * SEQ * HDIM;
    const float* kh = k + head * SEQ * HDIM;
    int m0 = mt * 16, n0 = nt * 16;

    // A: q[qi][d] (sI=64,sK=1).  B element (d,kj) = kh[kj*64+d] (sI=64,sK=1).
    v8f c = mma_tile<HDIM>(qh, HDIM, 1, m0, kh, HDIM, 1, n0, lane);

    int n = lane & 15, half = lane >> 4;
    int kj = n0 + n;
#pragma unroll
    for (int vv = 0; vv < 8; ++vv) {
        int qi = m0 + vv + 8 * half;
        float val = c[vv] + rel_bias[head * SEQ * SEQ + kj * SEQ + qi];
        attn_w[head * SEQ * SEQ + qi * SEQ + kj] = val;
    }
}

// ---------------------------------------------------------------------------
// Kernel 5: row softmax over keys.  One 128-thread block per (head,qi) row.
// ---------------------------------------------------------------------------
__global__ void softmax_kernel(const float* __restrict__ attn_w,
                               float* __restrict__ probs) {
    int row = blockIdx.x;                 // head*SEQ + qi
    int tid = threadIdx.x;
    float x = attn_w[row * SEQ + tid];

    float m = x;
    for (int off = 16; off; off >>= 1) m = fmaxf(m, __shfl_xor(m, off, 32));
    __shared__ float red[4];
    if ((tid & 31) == 0) red[tid >> 5] = m;
    __syncthreads();
    float bm = fmaxf(fmaxf(red[0], red[1]), fmaxf(red[2], red[3]));

    float e = expf(x - bm);
    float s = e;
    for (int off = 16; off; off >>= 1) s += __shfl_xor(s, off, 32);
    __syncthreads();
    if ((tid & 31) == 0) red[tid >> 5] = s;
    __syncthreads();
    float bs = red[0] + red[1] + red[2] + red[3];

    probs[row * SEQ + tid] = e / bs;
}

// ---------------------------------------------------------------------------
// Kernel 6: attn_output[h] = probs[h] @ v[h].  256 waves.
// ---------------------------------------------------------------------------
__global__ void attnout_kernel(const float* __restrict__ probs,
                               const float* __restrict__ v,
                               float* __restrict__ out) {
    int wave = (blockIdx.x * blockDim.x + threadIdx.x) >> 5;
    int lane = threadIdx.x & 31;
    int head = wave >> 5;
    int t    = wave & 31;
    int mt   = t >> 2, nt = t & 3;
    const float* ph = probs + head * SEQ * SEQ;
    const float* vh = v + head * SEQ * HDIM;
    int m0 = mt * 16, n0 = nt * 16;

    // A: probs[qi][kj] (sI=128,sK=1).  B element (kj,d)=vh[kj*64+d] (sI=1,sK=64).
    v8f c = mma_tile<SEQ>(ph, SEQ, 1, m0, vh, 1, HDIM, n0, lane);

    int n = lane & 15, half = lane >> 4;
#pragma unroll
    for (int vv = 0; vv < 8; ++vv) {
        int qi = m0 + vv + 8 * half;
        out[head * SEQ * HDIM + qi * HDIM + n0 + n] = c[vv];
    }
}

// ---------------------------------------------------------------------------
extern "C" void kernel_launch(void* const* d_in, const int* in_sizes, int n_in,
                              void* d_out, int out_size, void* d_ws, size_t ws_size,
                              hipStream_t stream) {
    const int*   x      = (const int*)d_in[0];
    const int*   x_len  = (const int*)d_in[1];
    const float* embed  = (const float*)d_in[2];
    const float* scale  = (const float*)d_in[3];
    const float* rbw    = (const float*)d_in[4];
    const float* qw     = (const float*)d_in[5];
    const float* kw     = (const float*)d_in[6];
    const float* vw     = (const float*)d_in[7];

    float* out      = (float*)d_out;
    float* rel_bias = out;                         // 8*128*128
    float* attn_w   = out + NHEADS * SEQ * SEQ;    // 8*128*128
    float* attn_o   = out + 2 * NHEADS * SEQ * SEQ;// 8*128*64

    // Workspace (floats). probs overlays h/q which are dead by then.
    float* ws = (float*)d_ws;
    float* h  = ws;                        // 128*512          = 65536
    float* q  = ws + 65536;                // 8*128*64         = 65536
    float* k  = ws + 131072;               // 8*128*64
    float* v  = ws + 196608;               // 8*128*64
    float* pr = ws;                        // 8*128*128 = 131072 (overlays h,q)

    relbias_kernel<<<(NHEADS * SEQ * SEQ + 255) / 256, 256, 0, stream>>>(rbw, x_len, rel_bias);
    rmsnorm_kernel<<<SEQ, 128, 0, stream>>>(x, x_len, embed, scale, h);
    qkv_kernel<<<96, 256, 0, stream>>>(h, qw, kw, vw, q, k, v);       // 768 waves
    scores_kernel<<<128, 128, 0, stream>>>(q, k, rel_bias, attn_w);   // 512 waves
    softmax_kernel<<<NHEADS * SEQ, 128, 0, stream>>>(attn_w, pr);
    attnout_kernel<<<64, 128, 0, stream>>>(pr, v, attn_o);            // 256 waves
}
